// ProbSparseAttention_4982162063454
// MI455X (gfx1250) — matrix operand order repaired and checked by reading, compiled, not run on hardware
//
#include <hip/hip_runtime.h>
#include <hip/hip_bf16.h>
#include <stdint.h>

// Problem constants (from reference): B=2, N=2048, D=1024, H=16, dh=64
#define BATCH   2
#define SEQ     2048
#define DMODEL  1024
#define NHEADS  16
#define DH      64
#define KSPARSE 204            // int(2048 * 0.1)

typedef __attribute__((ext_vector_type(16))) __bf16 v16bf;
typedef __attribute__((ext_vector_type(8)))  float  v8f;
typedef __attribute__((ext_vector_type(4)))  unsigned v4u;
typedef __attribute__((ext_vector_type(8)))  unsigned v8u;

union FragBF {                 // 16 bf16 = 8 VGPRs = one WMMA A/B operand
    v16bf v;
    uint4 q[2];
};

__device__ __forceinline__ unsigned short f2bf(float f) {
    unsigned int u = __float_as_uint(f);
    u += 0x7FFFu + ((u >> 16) & 1u);   // round-to-nearest-even
    return (unsigned short)(u >> 16);
}

// ---------------------------------------------------------------------------
// Tensor Data Mover: async 2D tile load Global -> LDS (CDNA5 TDM, TENSORcnt).
// D# per CDNA5 ISA. Issue once per workgroup (one wave);
// EXEC is ignored by TDM. dims/strides in data_size units (bf16 = 2 bytes).
// LDS destination receives tile rows contiguously: [tile_d1][tile_d0].
// ---------------------------------------------------------------------------
__device__ __forceinline__ void tdm_load_2d_bf16(const void* lds_dst,
                                                 const void* gsrc,
                                                 unsigned tensor_d0, unsigned tensor_d1,
                                                 unsigned tile_d0,   unsigned tile_d1,
                                                 unsigned stride0) {
    unsigned long long ga = (unsigned long long)(uintptr_t)gsrc;
    unsigned lds_off = (unsigned)(uintptr_t)lds_dst;   // flat low 32 bits = LDS offset
    v4u g0;
    g0[0] = 1u;                                        // count=1, user descriptor
    g0[1] = lds_off;                                   // lds_addr (bytes)
    g0[2] = (unsigned)ga;                              // global_addr[31:0]
    g0[3] = ((unsigned)(ga >> 32) & 0x01FFFFFFu) | (2u << 30);  // addr[56:32] | type=2
    v8u g1;
    g1[0] = (1u << 16);                                // wg_mask=0, data_size=1 (2B)
    g1[1] = (tensor_d0 << 16);                         // tensor_dim0[15:0] << 16
    g1[2] = (tensor_d0 >> 16) | (tensor_d1 << 16);     // dim0 hi | dim1 lo
    g1[3] = (tensor_d1 >> 16) | (tile_d0 << 16);       // dim1 hi | tile_dim0
    g1[4] = tile_d1 & 0xFFFFu;                         // tile_dim1 (tile_dim2=0)
    g1[5] = stride0;                                   // tensor_dim0_stride lo32
    g1[6] = 0u;                                        // stride0 hi | stride1 lo (2D)
    g1[7] = 0u;
    asm volatile("tensor_load_to_lds %0, %1" :: "s"(g0), "s"(g1) : "memory");
}

// ---------------------------------------------------------------------------
// Kernel 1: fp32 -> bf16 (vectorized x4)
// ---------------------------------------------------------------------------
__global__ void cvt_f32_bf16(const float* __restrict__ src,
                             unsigned short* __restrict__ dst, int n4) {
    int i = blockIdx.x * blockDim.x + threadIdx.x;
    if (i < n4) {
        float4 f = ((const float4*)src)[i];
        ushort4 o;
        o.x = f2bf(f.x); o.y = f2bf(f.y); o.z = f2bf(f.z); o.w = f2bf(f.w);
        ((ushort4*)dst)[i] = o;
    }
}

// ---------------------------------------------------------------------------
// Kernels 2 & 4: WMMA GEMM  C[M,N] = A[M,K] * B[N,K]^T + bias[N]
// Block tile 64(M) x 128(N), k-step 32, 256 threads = 8 waves.
// Double-buffered TDM pipeline: wave 0 issues D#s for k-tile i+1 into the
// alternate LDS buffer, then s_wait_tensorcnt 2 (TENSORcnt retires in issue
// order, so <=2 outstanding means tile i has landed) while all waves WMMA on
// tile i. Barriers order buffer reuse against the DMA overwrite.
// EPI==0: scatter bf16 into q(scaled by 1/sqrt(dh))/k/vT  (+bias)
// EPI==1: fp32 row-major output (+bias)
// ---------------------------------------------------------------------------
template <int EPI>
__global__ void __launch_bounds__(256)
gemm_bf16_wmma(const unsigned short* __restrict__ A,
               const unsigned short* __restrict__ Bm,
               const float* __restrict__ bias,
               unsigned short* __restrict__ q_out,
               unsigned short* __restrict__ k_out,
               unsigned short* __restrict__ vT_out,
               float* __restrict__ f32_out,
               int M, int N, int K) {
    __shared__ unsigned short lds_a[2 * 64 * 32];      // double-buffered A tile
    __shared__ unsigned short lds_b[2 * 128 * 32];     // double-buffered B tile

    const int m0 = blockIdx.y * 64;
    const int n0 = blockIdx.x * 128;
    const int t  = threadIdx.x;
    const int w  = t >> 5;
    const int L  = t & 31;
    const int wm = (w & 3) * 16;
    const int wn = (w >> 2) * 64;

    v8f acc[4] = {};

    // prologue: DMA k-tile 0 into buffer 0
    if (w == 0) {
        tdm_load_2d_bf16(&lds_a[0], &A[(size_t)m0 * K],
                         (unsigned)K, (unsigned)M, 32u, 64u, (unsigned)K);
        tdm_load_2d_bf16(&lds_b[0], &Bm[(size_t)n0 * K],
                         (unsigned)K, (unsigned)N, 32u, 128u, (unsigned)K);
    }

    for (int k0 = 0; k0 < K; k0 += 32) {
        const int cur = (k0 >> 5) & 1;
        if (w == 0) {
            if (k0 + 32 < K) {
                const int nxt = cur ^ 1;
                tdm_load_2d_bf16(&lds_a[nxt * 64 * 32],
                                 &A[(size_t)m0 * K + k0 + 32],
                                 (unsigned)K, (unsigned)M, 32u, 64u, (unsigned)K);
                tdm_load_2d_bf16(&lds_b[nxt * 128 * 32],
                                 &Bm[(size_t)n0 * K + k0 + 32],
                                 (unsigned)K, (unsigned)N, 32u, 128u, (unsigned)K);
                __builtin_amdgcn_s_wait_tensorcnt(2);  // tile k0 landed; k0+32 in flight
            } else {
                __builtin_amdgcn_s_wait_tensorcnt(0);  // drain on last tile
            }
        }
        __syncthreads();    // publish TDM-written LDS tile to all waves
        const unsigned short* la = &lds_a[cur * 64 * 32];
        const unsigned short* lb = &lds_b[cur * 128 * 32];

        // A fragment: lane row = wm + (L&15); elems 0-15 <- K base..base+15,
        // base = (L>>4)*16
        FragBF fa;
        {
            int row  = wm + (L & 15);
            int base = (L >> 4) * 16;
            fa.q[0] = *(const uint4*)&la[row * 32 + base];
            fa.q[1] = *(const uint4*)&la[row * 32 + base + 8];
        }
#pragma unroll
        for (int j = 0; j < 4; ++j) {
            // B fragment: lane col = n + (L&15); elems i <- K (L>>4)*16 + i
            FragBF fb;
            int n  = wn + j * 16 + (L & 15);
            int kb = (L >> 4) * 16;
            fb.q[0] = *(const uint4*)&lb[n * 32 + kb];
            fb.q[1] = *(const uint4*)&lb[n * 32 + kb + 8];
            acc[j] = __builtin_amdgcn_wmma_f32_16x16x32_bf16(
                false, fa.v, false, fb.v, (short)0, acc[j], false, false);
        }
        __syncthreads();    // reads done before next iteration's DMA overwrite
    }

    // Epilogue. C layout: VGPR r -> row r + 8*(L>=16), col = L&15.
#pragma unroll
    for (int j = 0; j < 4; ++j) {
#pragma unroll
        for (int r = 0; r < 8; ++r) {
            int row = m0 + wm + r + ((L >> 4) * 8);
            int col = n0 + wn + j * 16 + (L & 15);
            float val = acc[j][r] + bias[col];
            if (EPI == 0) {
                int sec = col >> 10;           // 0:q 1:k 2:v
                int h   = (col & 1023) >> 6;
                int d   = col & 63;
                int b   = row >> 11;
                int nq  = row & 2047;
                size_t hb = (size_t)(b * NHEADS + h);
                if (sec == 0) {
                    // fold 1/sqrt(dh) into Q so the attention kernel skips the
                    // post-WMMA v_pk_mul (and its wmma->valu hazard nops)
                    q_out[(hb * SEQ + nq) * DH + d] = f2bf(val * 0.125f);
                } else if (sec == 1) {
                    k_out[(hb * SEQ + nq) * DH + d] = f2bf(val);
                } else {
                    vT_out[(hb * DH + d) * SEQ + nq] = f2bf(val);
                }
            } else {
                f32_out[(size_t)row * N + col] = val;
            }
        }
    }
}

// ---------------------------------------------------------------------------
// Kernel 3: fused ProbSparse attention.
// One workgroup = one (b, h, 16-query-row block). 256 threads = 8 waves.
// Dynamic LDS: scores f32 [16][2048] (128KB) + P bf16 [16][2048] (64KB).
// CDNA5 WGP has 320KB LDS, so 192KB/workgroup is legal.
// ---------------------------------------------------------------------------
__global__ void __launch_bounds__(256)
probsparse_attn(const unsigned short* __restrict__ qg,
                const unsigned short* __restrict__ kg,
                const unsigned short* __restrict__ vTg,
                unsigned short* __restrict__ og) {
    extern __shared__ char smem[];
    float*          s_lds = (float*)smem;                          // [16][SEQ]
    unsigned short* p_lds = (unsigned short*)(smem + 16 * SEQ * 4);// [16][SEQ]
    float*          red   = (float*)smem;                          // reuse after P built

    const int n0 = blockIdx.x * 16;            // query-row block
    const int h  = blockIdx.y;
    const int b  = blockIdx.z;
    const int t  = threadIdx.x;
    const int w  = t >> 5;
    const int L  = t & 31;

    const unsigned short* qh = qg  + (size_t)(b * NHEADS + h) * SEQ * DH;
    const unsigned short* kh = kg  + (size_t)(b * NHEADS + h) * SEQ * DH;
    const unsigned short* vh = vTg + (size_t)(b * NHEADS + h) * DH * SEQ;

    // ---- Phase 1: S = (Q/8) K^T into LDS, WMMA over dh=64 (2 k-steps) ----
    FragBF fq[2];
    {
        int row  = n0 + (L & 15);
        int base = (L >> 4) * 16;
        const unsigned short* qrow = qh + (size_t)row * DH;
        fq[0].q[0] = *(const uint4*)&qrow[base];
        fq[0].q[1] = *(const uint4*)&qrow[base + 8];
        fq[1].q[0] = *(const uint4*)&qrow[32 + base];
        fq[1].q[1] = *(const uint4*)&qrow[32 + base + 8];
    }
    for (int nt = w; nt < SEQ / 16; nt += 8) {
        v8f acc = {};
#pragma unroll
        for (int ks = 0; ks < 2; ++ks) {
            FragBF fk;
            int key = nt * 16 + (L & 15);
            int kb  = ks * 32 + (L >> 4) * 16;
            const unsigned short* kr = kh + (size_t)key * DH + kb;
            fk.q[0] = *(const uint4*)&kr[0];
            fk.q[1] = *(const uint4*)&kr[8];
            acc = __builtin_amdgcn_wmma_f32_16x16x32_bf16(
                false, fq[ks].v, false, fk.v, (short)0, acc, false, false);
        }
#pragma unroll
        for (int r = 0; r < 8; ++r) {
            int row = r + ((L >> 4) * 8);
            int col = nt * 16 + (L & 15);
            s_lds[row * SEQ + col] = acc[r];   // scale pre-folded into Q
        }
    }
    __syncthreads();

    // ---- Phase 2: per-row top-k threshold (bisection) + masked softmax ----
    for (int rr = 0; rr < 2; ++rr) {
        int row = w * 2 + rr;                  // 8 waves x 2 rows = 16 rows
        float vals[64];
#pragma unroll
        for (int j = 0; j < 64; ++j) vals[j] = s_lds[row * SEQ + L + 32 * j];

        float mx = -__builtin_inff(), mn = __builtin_inff();
#pragma unroll
        for (int j = 0; j < 64; ++j) { mx = fmaxf(mx, vals[j]); mn = fminf(mn, vals[j]); }
#pragma unroll
        for (int o = 16; o > 0; o >>= 1) {
            mx = fmaxf(mx, __shfl_xor(mx, o, 32));
            mn = fminf(mn, __shfl_xor(mn, o, 32));
        }

        // bisect for the k-th largest value: largest t with count(>=t) >= k
        float lo = mn, hi = mx;
        for (int it = 0; it < 24; ++it) {
            float mid = 0.5f * (lo + hi);
            int c = 0;
#pragma unroll
            for (int j = 0; j < 64; ++j) c += (vals[j] >= mid) ? 1 : 0;
#pragma unroll
            for (int o = 16; o > 0; o >>= 1) c += __shfl_xor(c, o, 32);
            if (c >= KSPARSE) lo = mid; else hi = mid;
        }
        float th = lo;

        float sum = 0.f;
#pragma unroll
        for (int j = 0; j < 64; ++j)
            if (vals[j] >= th) sum += __expf(vals[j] - mx);
#pragma unroll
        for (int o = 16; o > 0; o >>= 1) sum += __shfl_xor(sum, o, 32);
        float inv = 1.f / sum;

#pragma unroll
        for (int j = 0; j < 64; ++j) {
            float p = (vals[j] >= th) ? __expf(vals[j] - mx) * inv : 0.f;
            p_lds[row * SEQ + L + 32 * j] = f2bf(p);
        }
    }
    __syncthreads();

    // ---- Phase 3: O = P @ V, split-K across wave pairs, LDS reduce ----
    const int ntile = w >> 1;                  // dh tile 0..3
    const int half  = w & 1;                   // K half
    v8f acc = {};
    for (int k0 = half * (SEQ / 2); k0 < (half + 1) * (SEQ / 2); k0 += 32) {
        FragBF fp, fv;
        {
            int row  = (L & 15);
            int base = (L >> 4) * 16;
            fp.q[0] = *(const uint4*)&p_lds[row * SEQ + k0 + base];
            fp.q[1] = *(const uint4*)&p_lds[row * SEQ + k0 + base + 8];
        }
        {
            int d  = ntile * 16 + (L & 15);
            int kb = k0 + (L >> 4) * 16;
            const unsigned short* vr = vh + (size_t)d * SEQ + kb;
            fv.q[0] = *(const uint4*)&vr[0];
            fv.q[1] = *(const uint4*)&vr[8];
        }
        acc = __builtin_amdgcn_wmma_f32_16x16x32_bf16(
            false, fp.v, false, fv.v, (short)0, acc, false, false);
    }
    __syncthreads();    // score region dead -> reuse as reduction scratch
    if (half == 1) {
#pragma unroll
        for (int r = 0; r < 8; ++r) {
            int row = r + ((L >> 4) * 8);
            int col = (L & 15);
            red[ntile * 256 + row * 16 + col] = acc[r];
        }
    }
    __syncthreads();
    if (half == 0) {
#pragma unroll
        for (int r = 0; r < 8; ++r) {
            int row = r + ((L >> 4) * 8);
            int col = (L & 15);
            float o = acc[r] + red[ntile * 256 + row * 16 + col];
            og[((size_t)(b * SEQ + n0 + row)) * DMODEL + h * DH + ntile * 16 + col] = f2bf(o);
        }
    }
}

// ---------------------------------------------------------------------------
// Host orchestration
// ---------------------------------------------------------------------------
extern "C" void kernel_launch(void* const* d_in, const int* in_sizes, int n_in,
                              void* d_out, int out_size, void* d_ws, size_t ws_size,
                              hipStream_t stream) {
    const float* x     = (const float*)d_in[0];   // [2,2048,1024]
    const float* W_qkv = (const float*)d_in[1];   // [3072,1024]
    const float* b_qkv = (const float*)d_in[2];   // [3072]
    const float* W_out = (const float*)d_in[3];   // [1024,1024]
    const float* b_out = (const float*)d_in[4];   // [1024]

    const size_t M  = (size_t)BATCH * SEQ;        // 4096
    const size_t NX = M * DMODEL;                 // x elems
    const size_t NWQ = (size_t)3 * DMODEL * DMODEL;
    const size_t NWO = (size_t)DMODEL * DMODEL;
    const size_t HEADELEMS = (size_t)BATCH * NHEADS * SEQ * DH;   // per q/k/v

    // workspace layout (bf16 = 2 bytes each), 256B aligned
    char* ws = (char*)d_ws;
    size_t off = 0;
    auto take = [&](size_t bytes) { char* p = ws + off; off += (bytes + 255) & ~(size_t)255; return p; };
    unsigned short* xb  = (unsigned short*)take(NX * 2);
    unsigned short* Wqb = (unsigned short*)take(NWQ * 2);
    unsigned short* Wob = (unsigned short*)take(NWO * 2);
    unsigned short* qw  = (unsigned short*)take(HEADELEMS * 2);
    unsigned short* kw  = (unsigned short*)take(HEADELEMS * 2);
    unsigned short* vTw = (unsigned short*)take(HEADELEMS * 2);
    unsigned short* aow = (unsigned short*)take(NX * 2);          // attn output, [4096][1024]
    (void)ws_size; (void)in_sizes; (void)n_in; (void)out_size;

    // 1) downconvert inputs to bf16
    cvt_f32_bf16<<<dim3((unsigned)(NX  / 4 / 256)), 256, 0, stream>>>(x,     xb,  (int)(NX  / 4));
    cvt_f32_bf16<<<dim3((unsigned)(NWQ / 4 / 256)), 256, 0, stream>>>(W_qkv, Wqb, (int)(NWQ / 4));
    cvt_f32_bf16<<<dim3((unsigned)(NWO / 4 / 256)), 256, 0, stream>>>(W_out, Wob, (int)(NWO / 4));

    // 2) QKV projection: [4096,3072] = xb * Wqb^T, scatter into q/k/vT (bf16)
    gemm_bf16_wmma<0><<<dim3(3 * DMODEL / 128, (unsigned)(M / 64)), 256, 0, stream>>>(
        xb, Wqb, b_qkv, qw, kw, vTw, nullptr, (int)M, 3 * DMODEL, DMODEL);

    // 3) fused probsparse attention
    const int SMEM_ATTN = 16 * SEQ * 4 + 16 * SEQ * 2;  // 196608 B (<320KB WGP LDS)
    (void)hipFuncSetAttribute((const void*)probsparse_attn,
                              hipFuncAttributeMaxDynamicSharedMemorySize, SMEM_ATTN);
    probsparse_attn<<<dim3(SEQ / 16, NHEADS, BATCH), 256, SMEM_ATTN, stream>>>(qw, kw, vTw, aow);

    // 4) output projection: d_out[4096,1024] = aow * Wob^T + b_out (fp32 out)
    gemm_bf16_wmma<1><<<dim3(DMODEL / 128, (unsigned)(M / 64)), 256, 0, stream>>>(
        aow, Wob, b_out, nullptr, nullptr, nullptr, (float*)d_out, (int)M, DMODEL, DMODEL);
}
